// HyperbolicGraphEncoderTL_8358006358202
// MI455X (gfx1250) — matrix operand order repaired and checked by reading, not compiled, not run on hardware
//
#include <hip/hip_runtime.h>
#include <hip/hip_bf16.h>
#include <stdint.h>

// ---------------------------------------------------------------------------
// Types for CDNA5 WMMA (wave32, 16x16x32 f16 -> f32 accumulate)
// ---------------------------------------------------------------------------
typedef __attribute__((ext_vector_type(16))) _Float16 v16h;
typedef __attribute__((ext_vector_type(8)))  _Float16 v8h;
typedef __attribute__((ext_vector_type(8)))  float    v8f;

// ---------------------------------------------------------------------------
// Convert B[K x Nc] (row-major, ld=ldb, fp32) -> Bh[Nc x Kpad] (f16, transposed,
// zero-padded in K). Makes B fragments two aligned 16B loads.
// ---------------------------------------------------------------------------
__global__ void convert_bT_f16_kernel(const float* __restrict__ B, int ldb,
                                      _Float16* __restrict__ Bh,
                                      int K, int Nc, int Kpad)
{
    const int idx = blockIdx.x * blockDim.x + threadIdx.x;   // linear over Nc*Kpad
    if (idx >= Nc * Kpad) return;
    const int c = idx / Kpad;
    const int k = idx - c * Kpad;
    const float v = (k < K) ? B[(size_t)k * ldb + c] : 0.0f;
    Bh[idx] = (_Float16)v;
}

// ---------------------------------------------------------------------------
// Strip WMMA GEMM (fast path): C[M x 128] = A[M x K] @ B[K x 128]
// Requirements: M % 16 == 0, lda % 4 == 0 (16B-aligned rows), Nc == 128.
// One wave computes a 16x128 strip: A fragment loaded once per 32-K step
// (4x b128), reused by 8 unrolled WMMAs. Bh is the f16 transposed B.
// ---------------------------------------------------------------------------
__global__ __launch_bounds__(256)
void wmma_gemm_strip_kernel(const float* __restrict__ A, int lda,
                            const _Float16* __restrict__ Bh, int Kpad,
                            float* __restrict__ C, int ldc,
                            int M, int K,
                            const float* __restrict__ bias,
                            const float* __restrict__ rowScale,
                            int do_relu)
{
    const int lane = threadIdx.x & 31;
    const int wave = (blockIdx.x * blockDim.x + threadIdx.x) >> 5;
    const int tilesM = M >> 4;
    if (wave >= tilesM) return;                 // wave-uniform exit

    const int row0   = wave << 4;
    const int r      = lane & 15;
    const int hi     = lane >> 4;               // 0 | 1
    const int aKbase = hi << 3;                 // 0 | 8   (A 16x32 f16 layout)
    const int bKbase = hi << 3;                 // 0 | 8   (B 32x16 f16 layout)
    const int arow   = row0 + r;
    const float* __restrict__ arowp = A + (size_t)arow * lda;

    v8f acc[8];
    #pragma unroll
    for (int t = 0; t < 8; ++t) acc[t] = (v8f){};

    const int Kfull = K & ~31;
    for (int k0 = 0; k0 < Kfull; k0 += 32) {
        const float* ap = arowp + k0 + aKbase;
        if (k0 + 544 < K) __builtin_prefetch(ap + 512, 0, 1);   // global_prefetch_b8
        const float4 x0 = ((const float4*)ap)[0];
        const float4 x1 = ((const float4*)ap)[1];
        const float4 x2 = ((const float4*)(ap + 16))[0];
        const float4 x3 = ((const float4*)(ap + 16))[1];
        v16h af;
        af[0]=(_Float16)x0.x;  af[1]=(_Float16)x0.y;  af[2]=(_Float16)x0.z;  af[3]=(_Float16)x0.w;
        af[4]=(_Float16)x1.x;  af[5]=(_Float16)x1.y;  af[6]=(_Float16)x1.z;  af[7]=(_Float16)x1.w;
        af[8]=(_Float16)x2.x;  af[9]=(_Float16)x2.y;  af[10]=(_Float16)x2.z; af[11]=(_Float16)x2.w;
        af[12]=(_Float16)x3.x; af[13]=(_Float16)x3.y; af[14]=(_Float16)x3.z; af[15]=(_Float16)x3.w;

        #pragma unroll
        for (int t = 0; t < 8; ++t) {
            const _Float16* bp = Bh + (size_t)(t * 16 + r) * Kpad + k0 + bKbase;
            const v8h blo = ((const v8h*)bp)[0];
            const v8h bhi = ((const v8h*)(bp + 16))[0];
            v16h bm;
            #pragma unroll
            for (int j = 0; j < 8; ++j) { bm[j] = blo[j]; bm[8 + j] = bhi[j]; }
            acc[t] = __builtin_amdgcn_wmma_f32_16x16x32_f16(
                         false, af, false, bm, (short)0, acc[t], false, false);
        }
    }

    if (Kfull < K) {   // single uniform tail step; Bh zero-padded beyond K
        v16h af;
        #pragma unroll
        for (int j = 0; j < 16; ++j) {
            const int k  = Kfull + aKbase + j + ((j >= 8) ? 8 : 0);
            const int kk = (k < K) ? k : (K - 1);     // clamped, unconditional load
            const float v = arowp[kk];
            af[j] = (_Float16)((k < K) ? v : 0.0f);
        }
        #pragma unroll
        for (int t = 0; t < 8; ++t) {
            const _Float16* bp = Bh + (size_t)(t * 16 + r) * Kpad + Kfull + bKbase;
            const v8h blo = ((const v8h*)bp)[0];
            const v8h bhi = ((const v8h*)(bp + 16))[0];
            v16h bm;
            #pragma unroll
            for (int j = 0; j < 8; ++j) { bm[j] = blo[j]; bm[8 + j] = bhi[j]; }
            acc[t] = __builtin_amdgcn_wmma_f32_16x16x32_f16(
                         false, af, false, bm, (short)0, acc[t], false, false);
        }
    }

    // ---- writeback: VGPR v -> M = row0 + v + hi*8, N = t*16 + r ----
    const int mbase = row0 + (hi << 3);
    #pragma unroll
    for (int t = 0; t < 8; ++t) {
        const int ncol = t * 16 + r;
        const float bs = bias ? bias[ncol] : 0.0f;
        #pragma unroll
        for (int v = 0; v < 8; ++v) {
            const int m = mbase + v;
            float val = acc[t][v] + bs;
            if (rowScale) val *= rowScale[m];
            if (do_relu)  val = fmaxf(val, 0.0f);
            C[(size_t)m * ldc + ncol] = val;
        }
    }
}

// ---------------------------------------------------------------------------
// General WMMA GEMM (branch-free clamped loads): C = op(A) @ B  (+bias/scale/relu)
// TA=true: A stored K x M (lda = M), accessed transposed.
// One wave per 16x16 tile; used for the small cluster-level GEMMs.
// ---------------------------------------------------------------------------
template <bool TA>
__global__ __launch_bounds__(256)
void wmma_gemm_gen_kernel(const float* __restrict__ A, int lda,
                          const float* __restrict__ B, int ldb,
                          float* __restrict__ C, int ldc,
                          int M, int Nc, int K,
                          const float* __restrict__ bias,
                          const float* __restrict__ rowScale,
                          int do_relu)
{
    const int lane   = threadIdx.x & 31;
    const int tilesN = (Nc + 15) >> 4;
    const int tilesM = (M + 15) >> 4;
    const int tileId = blockIdx.x * (blockDim.x >> 5) + (threadIdx.x >> 5);
    if (tileId >= tilesM * tilesN) return;      // wave-uniform exit

    const int tm = tileId / tilesN;
    const int tn = tileId - tm * tilesN;
    const int row0 = tm << 4;
    const int col0 = tn << 4;

    const int r      = lane & 15;
    const int hi     = lane >> 4;
    const int aKbase = hi << 3;
    const int bKbase = hi << 3;
    const int arow   = row0 + r;
    const int bcol   = col0 + r;
    const int rr     = (arow < M)  ? arow : 0;  // clamped indices: loads stay
    const int cc     = (bcol < Nc) ? bcol : 0;  // unconditional (no exec branching)

    v8f acc = {};
    for (int k0 = 0; k0 < K; k0 += 32) {
        v16h af, bm;
        #pragma unroll
        for (int j = 0; j < 16; ++j) {
            const int k  = k0 + aKbase + j + ((j >= 8) ? 8 : 0);
            const int kk = (k < K) ? k : 0;
            const float v = TA ? A[(size_t)kk * lda + rr]
                               : A[(size_t)rr * lda + kk];
            af[j] = (_Float16)(((arow < M) && (k < K)) ? v : 0.0f);
        }
        #pragma unroll
        for (int j = 0; j < 16; ++j) {
            const int k  = k0 + bKbase + j + ((j >= 8) ? 8 : 0);
            const int kk = (k < K) ? k : 0;
            const float v = B[(size_t)kk * ldb + cc];
            bm[j] = (_Float16)(((bcol < Nc) && (k < K)) ? v : 0.0f);
        }
        acc = __builtin_amdgcn_wmma_f32_16x16x32_f16(
                  false, af, false, bm, (short)0, acc, false, false);
    }

    const int ncol = col0 + r;
    if (ncol < Nc) {
        const float bs = bias ? bias[ncol] : 0.0f;
        #pragma unroll
        for (int v = 0; v < 8; ++v) {
            const int m = row0 + v + (hi << 3);
            if (m < M) {
                float val = acc[v] + bs;
                if (rowScale) val *= rowScale[m];
                if (do_relu)  val = fmaxf(val, 0.0f);
                C[(size_t)m * ldc + ncol] = val;
            }
        }
    }
}

// ---------------------------------------------------------------------------
// x[:,0] = sqrt(1 + sum(x[:,1:]^2)); one wave per row
// ---------------------------------------------------------------------------
__global__ void lorentz_time_kernel(float* __restrict__ x, int rows, int Dp1)
{
    const int wave = (blockIdx.x * blockDim.x + threadIdx.x) >> 5;
    const int lane = threadIdx.x & 31;
    if (wave >= rows) return;
    float* r = x + (size_t)wave * Dp1;
    float s = 0.0f;
    for (int j = 1 + lane; j < Dp1; j += 32) { const float v = r[j]; s += v * v; }
    #pragma unroll
    for (int m = 16; m >= 1; m >>= 1) s += __shfl_xor(s, m, 32);
    if (lane == 0) r[0] = sqrtf(1.0f + s);
}

// out[c] = 1/(relu(colsum(A[:,c])-1)+1); one block per column (LDS tree reduce)
__global__ void colsum_recip_kernel(const float* __restrict__ A, int rows, int cols,
                                    float* __restrict__ out)
{
    __shared__ float red[256];
    const int c = blockIdx.x;
    float s = 0.0f;
    for (int r = threadIdx.x; r < rows; r += blockDim.x) s += A[(size_t)r * cols + c];
    red[threadIdx.x] = s;
    __syncthreads();
    for (int off = blockDim.x >> 1; off > 0; off >>= 1) {
        if ((int)threadIdx.x < off) red[threadIdx.x] += red[threadIdx.x + off];
        __syncthreads();
    }
    if (threadIdx.x == 0) out[c] = 1.0f / (fmaxf(red[0] - 1.0f, 0.0f) + 1.0f);
}

// out[r] = 1/(relu(sum_j A[r,j]*cs[j] - 1)+1)
__global__ void rowsum_recip_kernel(const float* __restrict__ A, int rows, int cols,
                                    const float* __restrict__ cs, float* __restrict__ out)
{
    const int r = blockIdx.x * blockDim.x + threadIdx.x;
    if (r >= rows) return;
    float s = 0.0f;
    for (int j = 0; j < cols; ++j) s += A[(size_t)r * cols + j] * cs[j];
    out[r] = 1.0f / (fmaxf(s - 1.0f, 0.0f) + 1.0f);
}

// fadj = exp(-arccosh(clip(-minkdot,1+eps))) + I (tiny)
__global__ void affinity_kernel(const float* __restrict__ emb, float* __restrict__ out,
                                int m, int Dp1)
{
    const int j = threadIdx.x, i = threadIdx.y;
    if (i >= m || j >= m) return;
    const float* a = emb + (size_t)i * Dp1;
    const float* b = emb + (size_t)j * Dp1;
    float g = -a[0] * b[0];
    for (int k = 1; k < Dp1; ++k) g += a[k] * b[k];
    const float al = fmaxf(-g, 1.0f + 1e-6f);
    const float d  = logf(al + sqrtf(al * al - 1.0f));
    out[(size_t)i * m + j] = expf(-d) + ((i == j) ? 1.0f : 0.0f);
}

// sadj = relu(struct_adj - I*1e4) + I
__global__ void sadj_kernel(const float* __restrict__ a, float* __restrict__ out, int m)
{
    const int idx = blockIdx.x * blockDim.x + threadIdx.x;
    if (idx >= m * m) return;
    const int i = idx / m, j = idx - i * m;
    const float diag = (i == j) ? 1.0f : 0.0f;
    out[idx] = fmaxf(a[idx] - diag * 10000.0f, 0.0f) + diag;
}

// x = exp_map(x, w * log_map(x, msg)); one wave per row
__global__ void hyp_update_kernel(float* __restrict__ x, const float* __restrict__ msg,
                                  int rows, int Dp1, float w)
{
    const int wave = (blockIdx.x * blockDim.x + threadIdx.x) >> 5;
    const int lane = threadIdx.x & 31;
    if (wave >= rows) return;
    float* xr = x + (size_t)wave * Dp1;
    const float* mr = msg + (size_t)wave * Dp1;

    float xv[5], mv[5];
    float dxm = 0.0f, dmm = 0.0f, dxx = 0.0f;
    #pragma unroll
    for (int i = 0; i < 5; ++i) {
        const int j = lane + 32 * i;
        float xa = 0.0f, ma = 0.0f;
        if (j < Dp1) { xa = xr[j]; ma = mr[j]; }
        xv[i] = xa; mv[i] = ma;
        const float sgn = (j == 0) ? -1.0f : 1.0f;
        dxm += sgn * xa * ma;
        dmm += sgn * ma * ma;
        dxx += sgn * xa * xa;
    }
    #pragma unroll
    for (int m = 16; m >= 1; m >>= 1) {
        dxm += __shfl_xor(dxm, m, 32);
        dmm += __shfl_xor(dmm, m, 32);
        dxx += __shfl_xor(dxx, m, 32);
    }
    const float alpha = fmaxf(-dxm, 1.0f + 1e-6f);
    const float s     = sqrtf(alpha * alpha - 1.0f);
    const float d     = logf(alpha + s);                       // arccosh
    const float c     = w * d / s;                             // v = c*(msg - alpha*x)
    const float q     = dmm - 2.0f * alpha * dxm + alpha * alpha * dxx;
    const float vn    = sqrtf(fmaxf(c * c * q, 1e-6f));
    const float ch    = coshf(vn);
    const float shs   = sinhf(vn) / vn;
    #pragma unroll
    for (int i = 0; i < 5; ++i) {
        const int j = lane + 32 * i;
        if (j < Dp1) {
            const float v = c * (mv[i] - alpha * xv[i]);
            xr[j] = ch * xv[i] + shs * v;
        }
    }
}

// raw_feat[n] = concat(lane_tab, type_tab, length_tab, node_tab) lookups
__global__ void gather_kernel(const int* __restrict__ nodef, const int* __restrict__ typef,
                              const int* __restrict__ lenf, const int* __restrict__ lanef,
                              const float* __restrict__ ntab, const float* __restrict__ ttab,
                              const float* __restrict__ ltab, const float* __restrict__ ctab,
                              float* __restrict__ raw,
                              int CD, int TD, int LD, int ND)
{
    const int n = blockIdx.x;
    const int j = threadIdx.x;
    const int EU = CD + TD + LD + ND;
    if (j >= EU) return;
    float v;
    if      (j < CD)           v = ctab[(size_t)lanef[n] * CD + j];
    else if (j < CD + TD)      v = ttab[(size_t)typef[n] * TD + (j - CD)];
    else if (j < CD + TD + LD) v = ltab[(size_t)lenf[n] * LD + (j - CD - TD)];
    else                       v = ntab[(size_t)nodef[n] * ND + (j - CD - TD - LD)];
    raw[(size_t)n * EU + j] = v;
}

// ---------------------------------------------------------------------------
// Host side
// ---------------------------------------------------------------------------
extern "C" void kernel_launch(void* const* d_in, const int* in_sizes, int n_in,
                              void* d_out, int out_size, void* d_ws, size_t ws_size,
                              hipStream_t stream)
{
    const int*   node_f = (const int*)d_in[0];
    const int*   type_f = (const int*)d_in[1];
    const int*   len_f  = (const int*)d_in[2];
    const int*   lane_f = (const int*)d_in[3];
    const float* adj    = (const float*)d_in[4];
    const float* sA     = (const float*)d_in[5];
    const float* sAs    = (const float*)d_in[6];   // struct_assign (N x M1)
    const float* fAs    = (const float*)d_in[7];   // fnc_assign    (M1 x M2)
    const float* ntab   = (const float*)d_in[8];
    const float* ttab   = (const float*)d_in[9];
    const float* ltab   = (const float*)d_in[10];
    const float* ctab   = (const float*)d_in[11];
    const float* hypW   = (const float*)d_in[12];
    const float* hypB   = (const float*)d_in[13];

    const int N   = in_sizes[0];
    const int M1  = in_sizes[6] / N;
    const int M2  = in_sizes[7] / M1;
    const int D   = in_sizes[13];
    const int EU  = in_sizes[12] / D;
    const int ND  = in_sizes[8] / 10000;
    const int TD  = in_sizes[9] / 20;
    const int LD  = in_sizes[10] / 200;
    const int CD  = in_sizes[11] / 10;
    const int Dp1 = D + 1;
    const int KpadMax = (N + 31) & ~31;

    // ---- workspace carve ----
    float* w = (float*)d_ws;
    float* raw  = w;  w += (size_t)N * EU;
    float* hyp  = w;  w += (size_t)N * Dp1;
    float* tmpN = w;  w += (size_t)N * D;
    float* smsg = w;  w += (size_t)N * Dp1;
    float* rrs  = w;  w += N;
    float* se   = w;  w += (size_t)M1 * Dp1;
    float* fe   = w;  w += (size_t)M2 * Dp1;
    float* fe2  = w;  w += (size_t)M2 * Dp1;
    float* fmsg = w;  w += (size_t)M1 * Dp1;
    float* tmpA = w;  w += (size_t)M2 * D;
    float* tmpB = w;  w += (size_t)M1 * D;
    float* fadj = w;  w += (size_t)M2 * M2;
    float* sadj = w;  w += (size_t)M1 * M1;
    float* css  = w;  w += M1;
    float* csf  = w;  w += M2;
    float* rrf  = w;  w += M1;
    w += ((size_t)(16 - (((uintptr_t)w >> 2) & 15)) & 15);   // 64B align for Bh
    _Float16* Bh = (_Float16*)w;
    w += ((size_t)D * KpadMax + 1) / 2;                      // f16 buffer in float units
    if ((size_t)((char*)w - (char*)d_ws) > ws_size) return;

    auto gemm_gen = [&](const float* A, int lda, bool tA, const float* B, int ldb,
                        float* C, int ldc, int M, int Nc, int K,
                        const float* bias, const float* rscale, int relu) {
        const int tiles  = ((M + 15) / 16) * ((Nc + 15) / 16);
        const int blocks = (tiles + 7) / 8;
        if (tA)
            wmma_gemm_gen_kernel<true><<<blocks, 256, 0, stream>>>(
                A, lda, B, ldb, C, ldc, M, Nc, K, bias, rscale, relu);
        else
            wmma_gemm_gen_kernel<false><<<blocks, 256, 0, stream>>>(
                A, lda, B, ldb, C, ldc, M, Nc, K, bias, rscale, relu);
    };
    auto gemm_strip = [&](const float* A, int lda, const float* B, int ldb,
                          float* C, int ldc, int M, int K,
                          const float* bias, const float* rscale, int relu) {
        if (D != 128 || (M & 15) || (lda & 3)) {             // safety fallback
            gemm_gen(A, lda, false, B, ldb, C, ldc, M, D, K, bias, rscale, relu);
            return;
        }
        const int Kpad = (K + 31) & ~31;
        const int tot  = D * Kpad;
        convert_bT_f16_kernel<<<(tot + 255) / 256, 256, 0, stream>>>(B, ldb, Bh, K, D, Kpad);
        const int tilesM = M >> 4;
        wmma_gemm_strip_kernel<<<(tilesM + 7) / 8, 256, 0, stream>>>(
            A, lda, Bh, Kpad, C, ldc, M, K, bias, rscale, relu);
    };
    auto lorentz = [&](float* X, int rows) {
        lorentz_time_kernel<<<(rows + 7) / 8, 256, 0, stream>>>(X, rows, Dp1);
    };
    auto update = [&](float* X, const float* Msg, int rows, float wgt) {
        hyp_update_kernel<<<(rows + 7) / 8, 256, 0, stream>>>(X, Msg, rows, Dp1, wgt);
    };

    // ---- initial embedding: hyp = lorentz(raw @ hyp_W + hyp_b) ----
    gather_kernel<<<N, ((EU + 31) / 32) * 32, 0, stream>>>(node_f, type_f, len_f, lane_f,
                                                           ntab, ttab, ltab, ctab, raw,
                                                           CD, TD, LD, ND);
    gemm_strip(raw, EU, hypW, D, hyp + 1, Dp1, N, EU, hypB, nullptr, 0);
    lorentz(hyp, N);

    // ---- two tl_core layers ----
    for (int layer = 0; layer < 2; ++layer) {
        const int base = 14 + layer * 6;
        const float* fW = (const float*)d_in[base + 0];
        const float* fb = (const float*)d_in[base + 1];
        const float* sW = (const float*)d_in[base + 2];
        const float* sb = (const float*)d_in[base + 3];
        const float* nW = (const float*)d_in[base + 4];
        const float* nb = (const float*)d_in[base + 5];

        colsum_recip_kernel<<<M1, 256, 0, stream>>>(sAs, N, M1, css);
        colsum_recip_kernel<<<M2, 256, 0, stream>>>(fAs, M1, M2, csf);

        // struct_emb = lorentz(sa_norm.T @ hyp[:,1:])
        gemm_gen(sAs, M1, true, hyp + 1, Dp1, se + 1, Dp1, M1, D, N, nullptr, css, 0);
        lorentz(se, M1);
        // fnc_emb = lorentz(fa_norm.T @ struct_emb[:,1:])
        gemm_gen(fAs, M2, true, se + 1, Dp1, fe + 1, Dp1, M2, D, M1, nullptr, csf, 0);
        lorentz(fe, M2);

        // fadj ; fnc_emb = hyp_gcn(fnc_emb, fadj, fW, fb)
        affinity_kernel<<<dim3(1), dim3(32, 32), 0, stream>>>(fe, fadj, M2, Dp1);
        gemm_gen(fe + 1, Dp1, false, fW, D, tmpA, D, M2, D, D, fb, nullptr, 0);
        gemm_gen(fadj, M2, false, tmpA, D, fe2 + 1, Dp1, M2, D, M2, nullptr, nullptr, 1);
        lorentz(fe2, M2);

        // fnc_msg = distribute(...); struct_emb = update(struct_emb, fnc_msg, 0.15)
        rowsum_recip_kernel<<<(M1 + 255) / 256, 256, 0, stream>>>(fAs, M1, M2, csf, rrf);
        gemm_gen(fAs, M2, false, fe2 + 1, Dp1, fmsg + 1, Dp1, M1, D, M2, nullptr, rrf, 0);
        lorentz(fmsg, M1);
        update(se, fmsg, M1, 0.15f);

        // sadj ; struct_emb = hyp_gcn(struct_emb, sadj, sW, sb)
        sadj_kernel<<<(M1 * M1 + 255) / 256, 256, 0, stream>>>(sA, sadj, M1);
        gemm_gen(se + 1, Dp1, false, sW, D, tmpB, D, M1, D, D, sb, nullptr, 0);
        gemm_gen(sadj, M1, false, tmpB, D, se + 1, Dp1, M1, D, M1, nullptr, nullptr, 1);
        lorentz(se, M1);

        // struct_msg = distribute(struct_emb, struct_assign, sa_norm)  [strip fast path]
        rowsum_recip_kernel<<<(N + 255) / 256, 256, 0, stream>>>(sAs, N, M1, css, rrs);
        gemm_strip(sAs, M1, se + 1, Dp1, smsg + 1, Dp1, N, M1, nullptr, rrs, 0);
        lorentz(smsg, N);
        update(hyp, smsg, N, 0.5f);

        // hyp = hyp_gcn(hyp, adj, nW, nb)  [dominant GEMM: adj streamed once]
        gemm_gen(hyp + 1, Dp1, false, nW, D, tmpN, D, N, D, D, nb, nullptr, 0);
        float* dst = (layer == 1) ? (float*)d_out : hyp;
        gemm_strip(adj, N, tmpN, D, dst + 1, Dp1, N, N, nullptr, nullptr, 1);
        lorentz(dst, N);
    }
}
